// ProperTRMv3_80590766342497
// MI455X (gfx1250) — matrix operand order repaired and checked by reading, not compiled, not run on hardware
//
#include <hip/hip_runtime.h>
#include <hip/hip_bf16.h>
#include <cstdint>

// ---------------- problem constants ----------------
#define B_SZ     8192
#define D_IN     256
#define HDIM     512
#define CDIM     10
#define INTERD   682
#define GU       1364          // 2*INTER
#define GU_P     1408          // padded to multiple of 128
#define INTER_P  704           // padded to multiple of 32
#define K_OUT    4
#define H_CYC    2
#define L_CYC    3

typedef __attribute__((ext_vector_type(16))) __bf16       v16bf;
typedef __attribute__((ext_vector_type(8)))  float        v8f;
typedef __attribute__((ext_vector_type(4)))  unsigned int v4u;
typedef __attribute__((ext_vector_type(4)))  float        v4f;

__device__ __forceinline__ unsigned short f2bf(float f) {
    unsigned int u = __builtin_bit_cast(unsigned int, f);
    u += 0x7FFFu + ((u >> 16) & 1u);                 // round-to-nearest-even
    return (unsigned short)(u >> 16);
}
__device__ __forceinline__ float bf2f(unsigned short h) {
    unsigned int u = ((unsigned int)h) << 16;
    return __builtin_bit_cast(float, u);
}
struct V16Pair { v4u lo; v4u hi; };
__device__ __forceinline__ v16bf mkfrag(v4u lo, v4u hi) {
    V16Pair p{lo, hi};
    return __builtin_bit_cast(v16bf, p);
}

// ---------------- WMMA GEMM ----------------
// C[M,N] = A[M,K] * B[K,N] (+bias). A bf16 row-major (stride K).
// BT bf16 = B transposed, [N,K] row-major (stride K).
// 256 threads = 8 waves (wave32); block tile 128x128; wave tile 64x32
// (4x2 v8f accumulators -> 8 v_wmma per K-step).
// A tile double-buffered through LDS (shared across the 4 N-waves of each
// M-half); B fragments direct from L2 (all weights are L2-resident on the
// 192 MB L2). One s_barrier per K-step; staging overlaps the WMMAs.
template <int N, int K, bool BF_OUT, bool HAS_BIAS>
__global__ __launch_bounds__(256) void wmma_gemm(
    const unsigned short* __restrict__ A,
    const unsigned short* __restrict__ BT,
    void* __restrict__ Cout,
    const float* __restrict__ bias)
{
    __shared__ unsigned short ldsA[2][128 * 32];   // 2 x 8 KB

    const int tid  = threadIdx.x;
    const int lane = tid & 31;
    const int wave = tid >> 5;
    const int wm = wave & 1;          // 2 waves along M
    const int wn = wave >> 1;         // 4 waves along N
    const int blockM = blockIdx.y * 128;
    const int blockN = blockIdx.x * 128;
    const int waveM  = wm * 64;                       // within block
    const int waveN  = blockN + wn * 32;              // global N base
    const int l16     = lane & 15;
    const int halfsel = lane >> 4;                    // 0 or 1

    // cooperative-staging chunk coordinates (2 x b128 per thread)
    const int r0 = tid >> 2,         o0 = (tid & 3) * 8;
    const int r1 = (tid + 256) >> 2, o1 = o0;         // (tid+256)&3 == tid&3

    auto stage = [&](int buf, int kk) {
        *(v4u*)&ldsA[buf][r0 * 32 + o0] =
            *(const v4u*)&A[(size_t)(blockM + r0) * K + kk + o0];
        *(v4u*)&ldsA[buf][r1 * 32 + o1] =
            *(const v4u*)&A[(size_t)(blockM + r1) * K + kk + o1];
    };

    v8f acc[4][2];
    #pragma unroll
    for (int i = 0; i < 4; ++i)
        #pragma unroll
        for (int j = 0; j < 2; ++j)
            #pragma unroll
            for (int e = 0; e < 8; ++e) acc[i][j][e] = 0.0f;

    stage(0, 0);
    __syncthreads();

    for (int kk = 0; kk < K; kk += 32) {
        const int buf = (kk >> 5) & 1;
        if (kk + 32 < K) stage(buf ^ 1, kk + 32);   // prefetch next tile

        // B fragments (BT[N][K]): lane = col N=l16, K kb..kb+15 contiguous
        v16bf bfrag[2];
        #pragma unroll
        for (int nt = 0; nt < 2; ++nt) {
            const unsigned short* bp =
                &BT[(size_t)(waveN + nt * 16 + l16) * K + kk + halfsel * 16];
            bfrag[nt] = mkfrag(*(const v4u*)bp, *(const v4u*)(bp + 8));
        }
        // A fragments from LDS: lane row M=l16, K kb..kb+15 contiguous
        v16bf afrag[4];
        #pragma unroll
        for (int mt = 0; mt < 4; ++mt) {
            const unsigned short* ap =
                &ldsA[buf][(waveM + mt * 16 + l16) * 32 + halfsel * 16];
            afrag[mt] = mkfrag(*(const v4u*)ap, *(const v4u*)(ap + 8));
        }

        #pragma unroll
        for (int mt = 0; mt < 4; ++mt)
            #pragma unroll
            for (int nt = 0; nt < 2; ++nt)
                acc[mt][nt] = __builtin_amdgcn_wmma_f32_16x16x32_bf16(
                    false, afrag[mt], false, bfrag[nt],
                    (short)0, acc[mt][nt], false, false);

        __syncthreads();   // next-tile stores visible; this tile's reads done
    }

    // epilogue: C/D layout -> VGPR r: lanes0-15 M=r, lanes16-31 M=8+r; N=l16
    #pragma unroll
    for (int mt = 0; mt < 4; ++mt) {
        #pragma unroll
        for (int nt = 0; nt < 2; ++nt) {
            const int n = waveN + nt * 16 + l16;
            float bv = 0.0f;
            if constexpr (HAS_BIAS) bv = bias[n];
            #pragma unroll
            for (int r = 0; r < 8; ++r) {
                const int m = blockM + waveM + mt * 16 + halfsel * 8 + r;
                const float v = acc[mt][nt][r] + bv;
                if constexpr (BF_OUT)
                    ((unsigned short*)Cout)[(size_t)m * N + n] = f2bf(v);
                else
                    ((float*)Cout)[(size_t)m * N + n] = v;
            }
        }
    }
}

// ---------------- helper kernels ----------------
// dst[Npad][Kpad] bf16 = transpose of src[K][N] f32, zero padded.
__global__ __launch_bounds__(256) void transpose_bf16(
    const float* __restrict__ src, unsigned short* __restrict__ dst,
    int K, int N, int Kpad, int Npad)
{
    size_t i = (size_t)blockIdx.x * 256 + threadIdx.x;
    size_t total = (size_t)Npad * Kpad;
    if (i >= total) return;
    int n = (int)(i / Kpad), k = (int)(i % Kpad);
    float v = (k < K && n < N) ? src[(size_t)k * N + n] : 0.0f;
    dst[i] = f2bf(v);
}

__global__ __launch_bounds__(256) void cast_bf16(
    const float* __restrict__ src, unsigned short* __restrict__ dst, int n)
{
    int i = blockIdx.x * 256 + threadIdx.x;
    if (i < n) dst[i] = f2bf(src[i]);
}

__global__ __launch_bounds__(256) void bcast_rows(
    const float* __restrict__ vec, float* __restrict__ out, int H, int n)
{
    int i = blockIdx.x * 256 + threadIdx.x;
    if (i < n) out[i] = vec[i % H];
}

__global__ __launch_bounds__(256) void fillz(float* __restrict__ p, int n)
{
    int i = blockIdx.x * 256 + threadIdx.x;
    if (i < n) p[i] = 0.0f;
}

// v = a + b (+c)(+d); write f32 and bf16 copies
__global__ __launch_bounds__(256) void add_cast(
    const float* __restrict__ a, const float* __restrict__ b,
    const float* __restrict__ c, const float* __restrict__ d,
    float* __restrict__ of, unsigned short* __restrict__ ob, int n)
{
    int i = blockIdx.x * 256 + threadIdx.x;
    if (i >= n) return;
    float v = a[i] + b[i];
    if (c) v += c[i];
    if (d) v += d[i];
    of[i] = v;
    ob[i] = f2bf(v);
}

// silu(gate)*up from act[B,GU_P] -> inter[B,INTER_P] (pad cols zeroed)
__global__ __launch_bounds__(256) void swiglu_k(
    const unsigned short* __restrict__ act, unsigned short* __restrict__ inter,
    int rows)
{
    int i = blockIdx.x * 256 + threadIdx.x;
    int total = rows * INTER_P;
    if (i >= total) return;
    int r = i / INTER_P, c = i % INTER_P;
    float v = 0.0f;
    if (c < INTERD) {
        float g = bf2f(act[(size_t)r * GU_P + c]);
        float u = bf2f(act[(size_t)r * GU_P + INTERD + c]);
        v = (g / (1.0f + __expf(-g))) * u;
    }
    inter[i] = f2bf(v);
}

// out = rms_norm(h + mlp); one wave per row of HDIM=512
__global__ __launch_bounds__(256) void rmsnorm_k(
    const float* __restrict__ h, const float* __restrict__ mlp,
    float* __restrict__ outf, unsigned short* __restrict__ outb, int rows)
{
    int wave = threadIdx.x >> 5, lane = threadIdx.x & 31;
    int row  = blockIdx.x * 8 + wave;
    const float* hp = h   + (size_t)row * HDIM;
    const float* mp = mlp + (size_t)row * HDIM;
    v4f t[4];
    float ss = 0.0f;
    #pragma unroll
    for (int i = 0; i < 4; ++i) {
        v4f av = *(const v4f*)(hp + lane * 16 + i * 4);
        v4f bv = *(const v4f*)(mp + lane * 16 + i * 4);
        v4f tv = av + bv;
        t[i] = tv;
        ss += tv[0]*tv[0] + tv[1]*tv[1] + tv[2]*tv[2] + tv[3]*tv[3];
    }
    #pragma unroll
    for (int off = 16; off >= 1; off >>= 1) ss += __shfl_xor(ss, off, 32);
    float sc = rsqrtf(ss * (1.0f / HDIM) + 1e-6f);
    #pragma unroll
    for (int i = 0; i < 4; ++i) {
        v4f o = t[i] * sc;
        size_t base = (size_t)row * HDIM + lane * 16 + i * 4;
        *(v4f*)(outf + base) = o;
        unsigned int p0 = (unsigned int)f2bf(o[0]) | ((unsigned int)f2bf(o[1]) << 16);
        unsigned int p1 = (unsigned int)f2bf(o[2]) | ((unsigned int)f2bf(o[3]) << 16);
        *(unsigned int*)&outb[base]     = p0;
        *(unsigned int*)&outb[base + 2] = p1;
    }
}

// logits = zH@W_out+b_out; softmax; feedback = probs@W_fb+b_fb. Wave per row.
__global__ __launch_bounds__(256) void head_k(
    const float* __restrict__ zH, const float* __restrict__ Wout,
    const float* __restrict__ bout, const float* __restrict__ Wfb,
    const float* __restrict__ bfb, float* __restrict__ fb,
    float* __restrict__ final_logits, float* __restrict__ slab, int rows)
{
    int wave = threadIdx.x >> 5, lane = threadIdx.x & 31;
    int row  = blockIdx.x * 8 + wave;
    const float* zp = zH + (size_t)row * HDIM;
    float acc[CDIM];
    #pragma unroll
    for (int c = 0; c < CDIM; ++c) acc[c] = 0.0f;
    for (int h = lane; h < HDIM; h += 32) {
        float z = zp[h];
        const float* w = Wout + (size_t)h * CDIM;
        #pragma unroll
        for (int c = 0; c < CDIM; ++c) acc[c] += z * w[c];
    }
    #pragma unroll
    for (int off = 16; off >= 1; off >>= 1)
        #pragma unroll
        for (int c = 0; c < CDIM; ++c) acc[c] += __shfl_xor(acc[c], off, 32);

    float lg[CDIM], mx = -3.4e38f;
    #pragma unroll
    for (int c = 0; c < CDIM; ++c) { lg[c] = acc[c] + bout[c]; mx = fmaxf(mx, lg[c]); }
    float s = 0.0f, p[CDIM];
    #pragma unroll
    for (int c = 0; c < CDIM; ++c) { p[c] = __expf(lg[c] - mx); s += p[c]; }
    float inv = 1.0f / s;
    #pragma unroll
    for (int c = 0; c < CDIM; ++c) p[c] *= inv;

    if (lane < CDIM) {
        slab[(size_t)row * CDIM + lane] = lg[lane];
        if (final_logits) final_logits[(size_t)row * CDIM + lane] = lg[lane];
    }
    for (int j = lane; j < HDIM; j += 32) {
        float f = bfb[j];
        #pragma unroll
        for (int c = 0; c < CDIM; ++c) f += p[c] * Wfb[(size_t)c * HDIM + j];
        fb[(size_t)row * HDIM + j] = f;
    }
}

// ---------------- host driver ----------------
extern "C" void kernel_launch(void* const* d_in, const int* in_sizes, int n_in,
                              void* d_out, int out_size, void* d_ws, size_t ws_size,
                              hipStream_t stream)
{
    const float* x    = (const float*)d_in[0];
    const float* W_in = (const float*)d_in[1];
    const float* b_in = (const float*)d_in[2];
    const float* W_fb = (const float*)d_in[3];
    const float* b_fb = (const float*)d_in[4];
    const float* Wgu1 = (const float*)d_in[5];
    const float* Wd1  = (const float*)d_in[6];
    const float* Wgu2 = (const float*)d_in[7];
    const float* Wd2  = (const float*)d_in[8];
    const float* Hini = (const float*)d_in[9];
    const float* Lini = (const float*)d_in[10];
    const float* Wout = (const float*)d_in[11];
    const float* bout = (const float*)d_in[12];
    float* out = (float*)d_out;

    char* ws = (char*)d_ws;
    size_t off = 0;
    auto alloc = [&](size_t bytes) -> char* {
        char* p = ws + off;
        off += (bytes + 255) & ~(size_t)255;
        return p;
    };
    unsigned short* wgu1T = (unsigned short*)alloc((size_t)GU_P * HDIM * 2);
    unsigned short* wgu2T = (unsigned short*)alloc((size_t)GU_P * HDIM * 2);
    unsigned short* wd1T  = (unsigned short*)alloc((size_t)HDIM * INTER_P * 2);
    unsigned short* wd2T  = (unsigned short*)alloc((size_t)HDIM * INTER_P * 2);
    unsigned short* winT  = (unsigned short*)alloc((size_t)HDIM * D_IN * 2);
    unsigned short* xbf   = (unsigned short*)alloc((size_t)B_SZ * D_IN * 2);
    float* xproj = (float*)alloc((size_t)B_SZ * HDIM * 4);
    float* zH    = (float*)alloc((size_t)B_SZ * HDIM * 4);
    float* zL    = (float*)alloc((size_t)B_SZ * HDIM * 4);
    float* hF    = (float*)alloc((size_t)B_SZ * HDIM * 4);
    float* mlp   = (float*)alloc((size_t)B_SZ * HDIM * 4);
    float* fb    = (float*)alloc((size_t)B_SZ * HDIM * 4);
    unsigned short* hB    = (unsigned short*)alloc((size_t)B_SZ * HDIM * 2);
    unsigned short* act1  = (unsigned short*)alloc((size_t)B_SZ * GU_P * 2);
    unsigned short* inter = (unsigned short*)alloc((size_t)B_SZ * INTER_P * 2);

    dim3 blk(256);
    auto nb = [](size_t n) { return (unsigned)((n + 255) / 256); };

    // weight conversion to transposed bf16 (padded)
    transpose_bf16<<<nb((size_t)GU_P * HDIM), blk, 0, stream>>>(Wgu1, wgu1T, HDIM, GU, HDIM, GU_P);
    transpose_bf16<<<nb((size_t)GU_P * HDIM), blk, 0, stream>>>(Wgu2, wgu2T, HDIM, GU, HDIM, GU_P);
    transpose_bf16<<<nb((size_t)HDIM * INTER_P), blk, 0, stream>>>(Wd1, wd1T, INTERD, HDIM, INTER_P, HDIM);
    transpose_bf16<<<nb((size_t)HDIM * INTER_P), blk, 0, stream>>>(Wd2, wd2T, INTERD, HDIM, INTER_P, HDIM);
    transpose_bf16<<<nb((size_t)HDIM * D_IN), blk, 0, stream>>>(W_in, winT, D_IN, HDIM, D_IN, HDIM);
    cast_bf16<<<nb((size_t)B_SZ * D_IN), blk, 0, stream>>>(x, xbf, B_SZ * D_IN);

    // x_proj = x @ W_in + b_in
    wmma_gemm<HDIM, D_IN, false, true><<<dim3(HDIM / 128, B_SZ / 128), blk, 0, stream>>>(
        xbf, winT, xproj, b_in);

    bcast_rows<<<nb((size_t)B_SZ * HDIM), blk, 0, stream>>>(Hini, zH, HDIM, B_SZ * HDIM);
    bcast_rows<<<nb((size_t)B_SZ * HDIM), blk, 0, stream>>>(Lini, zL, HDIM, B_SZ * HDIM);
    fillz<<<nb((size_t)B_SZ * HDIM), blk, 0, stream>>>(fb, B_SZ * HDIM);

    auto sublayer = [&](unsigned short* wguT, unsigned short* wdT, float* outf) {
        wmma_gemm<GU_P, HDIM, true, false><<<dim3(GU_P / 128, B_SZ / 128), blk, 0, stream>>>(
            hB, wguT, act1, nullptr);
        swiglu_k<<<nb((size_t)B_SZ * INTER_P), blk, 0, stream>>>(act1, inter, B_SZ);
        wmma_gemm<HDIM, INTER_P, false, false><<<dim3(HDIM / 128, B_SZ / 128), blk, 0, stream>>>(
            inter, wdT, mlp, nullptr);
        rmsnorm_k<<<B_SZ / 8, blk, 0, stream>>>(hF, mlp, outf, hB, B_SZ);
    };
    auto runblock = [&](float* zout) {
        sublayer(wgu1T, wd1T, hF);   // sublayer 1: h -> hF/hB
        sublayer(wgu2T, wd2T, zout); // sublayer 2: -> z state
    };

    const int nBH = B_SZ * HDIM;
    for (int k = 0; k < K_OUT; ++k) {
        for (int hc = 0; hc < H_CYC; ++hc) {
            for (int lc = 0; lc < L_CYC; ++lc) {
                // z_L = block(z_L + z_H + x_proj + feedback)
                add_cast<<<nb((size_t)nBH), blk, 0, stream>>>(zL, zH, xproj, fb, hF, hB, nBH);
                runblock(zL);
            }
            // z_H = block(z_H + z_L)
            add_cast<<<nb((size_t)nBH), blk, 0, stream>>>(zH, zL, nullptr, nullptr, hF, hB, nBH);
            runblock(zH);
        }
        float* slab = out + (size_t)B_SZ * CDIM + (size_t)k * B_SZ * CDIM;
        head_k<<<B_SZ / 8, blk, 0, stream>>>(
            zH, Wout, bout, W_fb, b_fb, fb,
            (k == K_OUT - 1) ? out : nullptr, slab, B_SZ);
    }
}
